// WLGNN_30090540875813
// MI455X (gfx1250) — matrix operand order repaired and checked by reading, not compiled, not run on hardware
//
#include <hip/hip_runtime.h>
#include <cstddef>

#define DLAT 64

typedef __attribute__((ext_vector_type(16))) _Float16 v16h;
typedef __attribute__((ext_vector_type(8)))  _Float16 v8h;   // 16B
typedef __attribute__((ext_vector_type(4)))  _Float16 v4h;   // 8B
typedef __attribute__((ext_vector_type(2)))  _Float16 h2;    // 4B
typedef __attribute__((ext_vector_type(8)))  float    v8f;

static __device__ __forceinline__ void atomAddF(float* p, float v) {
  (void)__hip_atomic_fetch_add(p, v, __ATOMIC_RELAXED, __HIP_MEMORY_SCOPE_AGENT);
}

// ---------------- embedding gather (f32 -> agg staging buffer) ----------------

__global__ void k_embed(const int* __restrict__ x, const float* __restrict__ table,
                        float* __restrict__ h, int n) {
  int idx = blockIdx.x * blockDim.x + threadIdx.x;      // over n*16 float4s
  if (idx >= n * (DLAT / 4)) return;
  int r = idx >> 4, q = idx & 15;
  ((float4*)h)[idx] = ((const float4*)(table + (size_t)x[r] * DLAT))[q];
}

// ---------------- degree / dinv ----------------

__global__ void k_deg_init(float* __restrict__ deg, int n) {
  int i = blockIdx.x * blockDim.x + threadIdx.x;
  if (i < n) deg[i] = 1.0f;            // self-loop contribution
}

__global__ void k_deg_count(const int* __restrict__ col, float* __restrict__ deg, int ne) {
  int i = blockIdx.x * blockDim.x + threadIdx.x;
  if (i < ne) atomAddF(&deg[col[i]], 1.0f);
}

__global__ void k_deg_rsqrt(float* __restrict__ deg, int n) {
  int i = blockIdx.x * blockDim.x + threadIdx.x;
  if (i < n) deg[i] = rsqrtf(deg[i]);
}

// ---------------- segment-sum: self term + edge scatter ----------------

// agg = xw * dinv^2   (xw stored f16, agg accumulates f32)
__global__ void k_self_init(const _Float16* __restrict__ xw, const float* __restrict__ dinv,
                            float* __restrict__ agg, int nrows) {
  int idx = blockIdx.x * blockDim.x + threadIdx.x;      // 4 elems per thread
  if (idx >= nrows * (DLAT / 4)) return;
  int r = idx >> 4;
  float di = dinv[r];
  float s = di * di;
  v4h v = ((const v4h*)xw)[idx];
  float4 o = {(float)v[0] * s, (float)v[1] * s, (float)v[2] * s, (float)v[3] * s};
  ((float4*)agg)[idx] = o;
}

// one wave per edge; lane gathers one dword (2 halves) of the 128B message row,
// accumulates in f32 via native global atomics (relaxed, agent scope)
__global__ void k_scatter(const _Float16* __restrict__ xw, const int* __restrict__ erow,
                          const int* __restrict__ ecol, const float* __restrict__ dinv,
                          float* __restrict__ agg, int ne) {
  int e = blockIdx.x * (blockDim.x >> 5) + (threadIdx.x >> 5);
  if (e >= ne) return;
  int lane = threadIdx.x & 31;
  int r = erow[e], c = ecol[e];
  float coeff = dinv[r] * dinv[c];
  h2 v = ((const h2*)(xw + (size_t)r * DLAT))[lane];
  float* dst = agg + (size_t)c * DLAT + 2 * lane;
  atomAddF(dst,     (float)v[0] * coeff);
  atomAddF(dst + 1, (float)v[1] * coeff);
}

// ---------------- GraphNorm: fused sum/sumsq reduction (f32 moments) ----------------
// stats[0:64]=sum, [64:128]=sumsq, [128:192]=shift(ms*mean), [192:256]=rsqrt(var+eps)

__global__ void k_zero_stats(float* __restrict__ stats) {
  stats[threadIdx.x] = 0.0f;
}

__global__ void k_gn_reduce(const float* __restrict__ X, const float* __restrict__ bias,
                            float* __restrict__ stats, int nrows) {
  int col = threadIdx.x;                       // blockDim.x == 64
  float b = bias ? bias[col] : 0.0f;
  float s = 0.0f, sq = 0.0f;
  for (int r = blockIdx.x; r < nrows; r += gridDim.x) {
    float v = X[(size_t)r * DLAT + col] + b;
    s += v; sq += v * v;
  }
  atomAddF(&stats[col], s);
  atomAddF(&stats[DLAT + col], sq);
}

// var of (x - ms*mean) = E[x^2] - (2ms - ms^2)*mean^2   (single-pass moments)
__global__ void k_gn_finalize(float* __restrict__ stats, const float* __restrict__ ms, float n) {
  int c = threadIdx.x;                         // one block, 64 threads
  float mean = stats[c] / n;
  float ex2  = stats[DLAT + c] / n;
  float m    = ms[c];
  float var  = ex2 - (2.0f * m - m * m) * mean * mean;
  stats[2 * DLAT + c] = m * mean;
  stats[3 * DLAT + c] = rsqrtf(var + 1e-5f);
}

// norm + bias + (relu) + (branch accumulate), f32 in -> f16 feature store
__global__ void k_gn_apply(const float* __restrict__ X, const float* __restrict__ bias,
                           const float* __restrict__ stats, const float* __restrict__ gw,
                           const float* __restrict__ gb, _Float16* __restrict__ Y,
                           int nrows, int relu, int accumulate) {
  int idx = blockIdx.x * blockDim.x + threadIdx.x;      // 4 elems per thread
  if (idx >= nrows * (DLAT / 4)) return;
  int q = idx & 15;                                     // float4 index within row
  float4 v  = ((const float4*)X)[idx];
  float4 sh = ((const float4*)(stats + 2 * DLAT))[q];
  float4 sc = ((const float4*)(stats + 3 * DLAT))[q];
  float4 w4 = ((const float4*)gw)[q];
  float4 b4 = ((const float4*)gb)[q];
  if (bias) {
    float4 bb = ((const float4*)bias)[q];
    v.x += bb.x; v.y += bb.y; v.z += bb.z; v.w += bb.w;
  }
  float4 y;
  y.x = w4.x * (v.x - sh.x) * sc.x + b4.x;
  y.y = w4.y * (v.y - sh.y) * sc.y + b4.y;
  y.z = w4.z * (v.z - sh.z) * sc.z + b4.z;
  y.w = w4.w * (v.w - sh.w) * sc.w + b4.w;
  if (relu) {
    y.x = fmaxf(y.x, 0.0f); y.y = fmaxf(y.y, 0.0f);
    y.z = fmaxf(y.z, 0.0f); y.w = fmaxf(y.w, 0.0f);
  }
  if (accumulate) {
    v4h o = ((const v4h*)Y)[idx];
    y.x += (float)o[0]; y.y += (float)o[1]; y.z += (float)o[2]; y.w += (float)o[3];
  }
  v4h out = {(_Float16)y.x, (_Float16)y.y, (_Float16)y.z, (_Float16)y.w};
  ((v4h*)Y)[idx] = out;
}

// ---------------- WMMA GEMM: C[nrows x 64](f16) = A[nrows x 64](f16) @ W[64 x 64] ----------
// 256 threads = 8 waves; each wave computes one 16x16 tile (2 m-tiles x 4 n-tiles / block).
// A stored f16: each lane's fragment = two contiguous 16B global_load_b128 (no cvt).
// W staged to LDS f16 column-major so each lane's B fragment is one contiguous 32B v16h.
// K=64 as two v_wmma_f32_16x16x32_f16 steps; f32 accumulate, f16 store.

__global__ void k_gemm64(const _Float16* __restrict__ A, const float* __restrict__ W,
                         _Float16* __restrict__ C, int nrows) {
  __shared__ _Float16 sWt[DLAT * DLAT];        // sWt[col*64 + k] = W[k][col]
  for (int i = threadIdx.x; i < DLAT * DLAT; i += blockDim.x) {
    int k = i >> 6, col = i & 63;
    sWt[col * DLAT + k] = (_Float16)W[i];
  }
  __syncthreads();

  int wave  = threadIdx.x >> 5;
  int lane  = threadIdx.x & 31;
  int mtile = wave >> 2;                 // 0..1
  int ntile = wave & 3;                  // 0..3
  int row0  = blockIdx.x * 32 + mtile * 16;
  int lrow  = lane & 15;
  int khalf = lane >> 4;
  int colc  = ntile * 16 + lrow;

  int arow = row0 + lrow;
  if (arow >= nrows) arow = nrows - 1;   // tail clamp (stores are guarded)
  const _Float16* ap = A + (size_t)arow * DLAT;

  v8f acc = {};
  #pragma unroll
  for (int k0 = 0; k0 < DLAT; k0 += 32) {
    // A 16x32 f16, this code's contiguous k order (not the ISA table order): lane<16 holds K 0..15, lane>=16 holds K 16..31 (matches B)
    v8h alo = *(const v8h*)(ap + k0 + 16 * khalf);       // K = 16kh .. 16kh+7
    v8h ahi = *(const v8h*)(ap + k0 + 16 * khalf + 8);   // K = 16kh+8 .. +7
    v16h af;
    #pragma unroll
    for (int e = 0; e < 8; ++e) { af[e] = alo[e]; af[8 + e] = ahi[e]; }
    // B 32x16 f16, same contiguous k order as A: lane<16 K 0..15, lane>=16 K 16..31, col = lrow -> contiguous in sWt
    v16h bf = *(const v16h*)(&sWt[colc * DLAT + k0 + 16 * khalf]);
    acc = __builtin_amdgcn_wmma_f32_16x16x32_f16(false, af, false, bf,
                                                 (short)0, acc, false, false);
  }

  // C f32 16x16 layout: lane = N + 16*(M>=8), vgpr v = M%8  -> store f16
  _Float16* cp = C + (size_t)(row0 + 8 * khalf) * DLAT + colc;
  if (row0 + 16 <= nrows) {              // wave-uniform fast path: no per-store guards
    #pragma unroll
    for (int v = 0; v < 8; ++v) cp[(size_t)v * DLAT] = (_Float16)acc[v];
  } else {
    #pragma unroll
    for (int v = 0; v < 8; ++v)
      if (row0 + 8 * khalf + v < nrows) cp[(size_t)v * DLAT] = (_Float16)acc[v];
  }
}

// ---------------- pair gathers + head (f16 features) ----------------

__global__ void k_pair_mul(const _Float16* __restrict__ h, const int* __restrict__ pos1,
                           _Float16* __restrict__ out, int m) {
  int idx = blockIdx.x * blockDim.x + threadIdx.x;      // 4 elems per thread
  if (idx >= m * (DLAT / 4)) return;
  int i = idx >> 4, q = idx & 15;
  int a = pos1[2 * i], b = pos1[2 * i + 1];
  v4h va = ((const v4h*)(h + (size_t)a * DLAT))[q];
  v4h vb = ((const v4h*)(h + (size_t)b * DLAT))[q];
  v4h r;
  #pragma unroll
  for (int e = 0; e < 4; ++e) r[e] = (_Float16)((float)va[e] * (float)vb[e]);
  ((v4h*)out)[idx] = r;
}

// wave-per-output: lanes cover 2 halves each of the two gathered rows, shfl reduce
__global__ void k_head(const _Float16* __restrict__ h, const int* __restrict__ pos2,
                       const float* __restrict__ pw, const float* __restrict__ pb,
                       float* __restrict__ out, int nout) {
  int j = blockIdx.x * (blockDim.x >> 5) + (threadIdx.x >> 5);
  if (j >= nout) return;
  int lane = threadIdx.x & 31;
  int a = pos2[2 * j], b = pos2[2 * j + 1];
  h2 va = ((const h2*)(h + (size_t)a * DLAT))[lane];
  h2 vb = ((const h2*)(h + (size_t)b * DLAT))[lane];
  float2 w = ((const float2*)pw)[lane];
  float p = (float)va[0] * (float)vb[0] * w.x + (float)va[1] * (float)vb[1] * w.y;
  #pragma unroll
  for (int off = 16; off > 0; off >>= 1) p += __shfl_down(p, off, 32);
  if (lane == 0) out[j] = p + pb[0];
}

// ---------------- host orchestration ----------------

static inline int cdiv(int a, int b) { return (a + b - 1) / b; }

extern "C" void kernel_launch(void* const* d_in, const int* in_sizes, int n_in,
                              void* d_out, int out_size, void* d_ws, size_t ws_size,
                              hipStream_t stream) {
  (void)n_in; (void)ws_size; (void)out_size;

  const int*   x       = (const int*)d_in[0];
  const int*   edge1   = (const int*)d_in[2];
  const int*   edge2   = (const int*)d_in[3];
  const int*   edge2r  = (const int*)d_in[4];
  const int*   pos1    = (const int*)d_in[5];
  const int*   pos2    = (const int*)d_in[6];
  const float* embT    = (const float*)d_in[7];
  const float* egw     = (const float*)d_in[8];
  const float* egb     = (const float*)d_in[9];
  const float* egms    = (const float*)d_in[10];
  const float* c1W  = (const float*)d_in[11]; const float* c1b  = (const float*)d_in[12];
  const float* c1gw = (const float*)d_in[13]; const float* c1gb = (const float*)d_in[14];
  const float* c1gms= (const float*)d_in[15];
  const float* c2W  = (const float*)d_in[16]; const float* c2b  = (const float*)d_in[17];
  const float* c2gw = (const float*)d_in[18]; const float* c2gb = (const float*)d_in[19];
  const float* c2gms= (const float*)d_in[20];
  const float* crW  = (const float*)d_in[21]; const float* crb  = (const float*)d_in[22];
  const float* crgw = (const float*)d_in[23]; const float* crgb = (const float*)d_in[24];
  const float* crgms= (const float*)d_in[25];
  const float* predW = (const float*)d_in[26];
  const float* predB = (const float*)d_in[27];

  const int N   = in_sizes[0];
  const int E1  = in_sizes[2] / 2;
  const int E2  = in_sizes[3] / 2;
  const int M   = in_sizes[5] / 2;
  const int NP2 = in_sizes[6];
  const int D1  = in_sizes[11] / (DLAT * DLAT);
  const int D2  = in_sizes[16] / (DLAT * DLAT);
  const int maxNM = (N > M) ? N : M;

  // workspace carve-up
  char* ws = (char*)d_ws;
  const size_t bigH = (size_t)maxNM * DLAT * sizeof(_Float16);
  const size_t bigF = (size_t)maxNM * DLAT * sizeof(float);
  _Float16* H0 = (_Float16*)ws; ws += bigH;   // feature ping
  _Float16* H1 = (_Float16*)ws; ws += bigH;   // feature pong
  _Float16* XW = (_Float16*)ws; ws += bigH;   // gemm output (messages)
  float*    AG = (float*)ws;    ws += bigF;   // f32 segment-sum accumulator / staging
  float* dinv1  = (float*)ws; ws += (size_t)N * sizeof(float);
  float* dinv2  = (float*)ws; ws += (size_t)M * sizeof(float);
  float* dinv2r = (float*)ws; ws += (size_t)M * sizeof(float);
  float* stats  = (float*)ws; ws += 4 * DLAT * sizeof(float);

  // degrees (once per edge list, reused across layers)
  k_deg_init<<<cdiv(N, 256), 256, 0, stream>>>(dinv1, N);
  k_deg_init<<<cdiv(M, 256), 256, 0, stream>>>(dinv2, M);
  k_deg_init<<<cdiv(M, 256), 256, 0, stream>>>(dinv2r, M);
  k_deg_count<<<cdiv(E1, 256), 256, 0, stream>>>(edge1 + E1, dinv1, E1);
  k_deg_count<<<cdiv(E2, 256), 256, 0, stream>>>(edge2 + E2, dinv2, E2);
  k_deg_count<<<cdiv(E2, 256), 256, 0, stream>>>(edge2r + E2, dinv2r, E2);
  k_deg_rsqrt<<<cdiv(N, 256), 256, 0, stream>>>(dinv1, N);
  k_deg_rsqrt<<<cdiv(M, 256), 256, 0, stream>>>(dinv2, M);
  k_deg_rsqrt<<<cdiv(M, 256), 256, 0, stream>>>(dinv2r, M);

  auto run_gn = [&](const float* X, const float* bias, const float* gw, const float* gb,
                    const float* gms, _Float16* Y, int nrows, bool relu, bool accum) {
    int rblocks = nrows < 1024 ? nrows : 1024;
    k_zero_stats<<<1, 4 * DLAT, 0, stream>>>(stats);
    k_gn_reduce<<<rblocks, DLAT, 0, stream>>>(X, bias, stats, nrows);
    k_gn_finalize<<<1, DLAT, 0, stream>>>(stats, gms, (float)nrows);
    k_gn_apply<<<cdiv(nrows * (DLAT / 4), 256), 256, 0, stream>>>(
        X, bias, stats, gw, gb, Y, nrows, relu ? 1 : 0, accum ? 1 : 0);
  };

  auto run_gcn = [&](const _Float16* hin, int nrows, const int* erow, const int* ecol, int ne,
                     const float* dinv, const float* W, const float* bias,
                     const float* gw, const float* gb, const float* gms,
                     _Float16* dest, bool relu, bool accum) {
    k_gemm64<<<cdiv(nrows, 32), 256, 0, stream>>>(hin, W, XW, nrows);
    k_self_init<<<cdiv(nrows * (DLAT / 4), 256), 256, 0, stream>>>(XW, dinv, AG, nrows);
    k_scatter<<<cdiv(ne, 8), 256, 0, stream>>>(XW, erow, ecol, dinv, AG, ne);
    run_gn(AG, bias, gw, gb, gms, dest, nrows, relu, accum);
  };

  // embedding (f32 staging in AG) + GraphNorm -> f16 features in H0
  k_embed<<<cdiv(N * (DLAT / 4), 256), 256, 0, stream>>>(x, embT, AG, N);
  run_gn(AG, nullptr, egw, egb, egms, H0, N, false, false);

  // conv1 stack on the N-node graph (gemm consumes H0 first, so in-place dest is safe)
  _Float16* cur = H0;
  for (int l = 0; l < D1; ++l) {
    run_gcn(cur, N, edge1, edge1 + E1, E1, dinv1,
            c1W + (size_t)l * DLAT * DLAT, c1b + l * DLAT,
            c1gw + l * DLAT, c1gb + l * DLAT, c1gms + l * DLAT,
            cur, true, false);
  }

  // pairwise gather-multiply -> M link-nodes (H0 -> H1)
  k_pair_mul<<<cdiv(M * (DLAT / 4), 256), 256, 0, stream>>>(cur, pos1, H1, M);
  _Float16* curM = H1;
  _Float16* spare = H0;

  // dual-branch conv2 stack on the M-node graph
  for (int l = 0; l < D2; ++l) {
    run_gcn(curM, M, edge2, edge2 + E2, E2, dinv2,
            c2W + (size_t)l * DLAT * DLAT, c2b + l * DLAT,
            c2gw + l * DLAT, c2gb + l * DLAT, c2gms + l * DLAT,
            spare, true, false);
    run_gcn(curM, M, edge2r, edge2r + E2, E2, dinv2r,
            crW + (size_t)l * DLAT * DLAT, crb + l * DLAT,
            crgw + l * DLAT, crgb + l * DLAT, crgms + l * DLAT,
            spare, true, true);               // accumulate: h = a + b2
    _Float16* t = curM; curM = spare; spare = t;
  }

  // pos2 gather, even/odd product, linear head
  int nout = NP2 / 2;
  k_head<<<cdiv(nout, 32), 256, 0, stream>>>(curM, pos2, predW, predB, (float*)d_out, nout);
}
